// DAGNN_19370302505614
// MI455X (gfx1250) — matrix operand order repaired and checked by reading, not compiled, not run on hardware
//
#include <hip/hip_runtime.h>
#include <hip/hip_bf16.h>

// ---------------- types for WMMA (wave32, gfx1250) ----------------
typedef __attribute__((ext_vector_type(16))) __bf16 v16bf;
typedef __attribute__((ext_vector_type(8)))  __bf16 bf16x8;
typedef __attribute__((ext_vector_type(8)))  float  v8f;

static __device__ __forceinline__ __bf16 f2bf(float f) { return (__bf16)f; }

#define F_IN   512
#define HID    64
#define NCLS   64

// =================================================================
// Kernel 1: logits = relu(X @ W1 + b1) @ W2 + b2   via v_wmma_f32_16x16x32_bf16
// block = 128 threads (4 waves), each wave computes a 16x64 tile.
// =================================================================
__global__ __launch_bounds__(128) void dagnn_mlp_wmma(
    const float* __restrict__ X,
    const float* __restrict__ W1, const float* __restrict__ b1,
    const float* __restrict__ W2, const float* __restrict__ b2,
    float* __restrict__ out, int n)
{
  __shared__ __bf16 w1t[HID * F_IN];     // W1 transposed [n][k], bf16 (64 KB)
  __shared__ __bf16 w2t[NCLS * HID];     // W2 transposed [n][k], bf16 (8 KB)
  __shared__ __bf16 hbuf[4][16 * HID];   // per-wave hidden tile, row-major [m][k]
  __shared__ float  b1s[HID], b2s[NCLS];

  const int tid = threadIdx.x;

  // Stage weights (transposed) into LDS as bf16.
  for (int i = tid; i < F_IN * HID; i += 128) {
    int k = i / HID, nn = i % HID;                 // W1 is [k][n] row-major
    w1t[nn * F_IN + k] = f2bf(W1[i]);
  }
  for (int i = tid; i < HID * NCLS; i += 128) {
    int k = i / NCLS, nn = i % NCLS;
    w2t[nn * HID + k] = f2bf(W2[i]);
  }
  if (tid < HID)  b1s[tid] = b1[tid];
  if (tid < NCLS) b2s[tid] = b2[tid];
  __syncthreads();

  const int wave = tid >> 5;
  const int lane = tid & 31;
  const int mlo  = lane & 15;   // row (or column) within 16
  const int hi   = lane >> 4;   // lane half selector

  long row0 = (long)blockIdx.x * 64 + (long)wave * 16;
  const bool valid = (row0 < n);
  if (row0 > (long)n - 16) row0 = (long)n - 16;   // clamp (n % 16 == 0)

  const float* Arow = X + (row0 + mlo) * F_IN;

  // ---------------- GEMM 1: 16x512 @ 512x64 ----------------
  v8f acc[4] = {v8f{}, v8f{}, v8f{}, v8f{}};
  for (int k0 = 0; k0 < F_IN; k0 += 32) {
    // A fragment (contiguous k = 16*hi + s, matching B)
    const int ka = k0 + hi * 16;
    float4 a0 = *(const float4*)(Arow + ka);
    float4 a1 = *(const float4*)(Arow + ka + 4);
    float4 a2 = *(const float4*)(Arow + ka + 8);
    float4 a3 = *(const float4*)(Arow + ka + 12);
    v16bf a;
    a[0]=f2bf(a0.x); a[1]=f2bf(a0.y); a[2]=f2bf(a0.z); a[3]=f2bf(a0.w);
    a[4]=f2bf(a1.x); a[5]=f2bf(a1.y); a[6]=f2bf(a1.z); a[7]=f2bf(a1.w);
    a[8]=f2bf(a2.x); a[9]=f2bf(a2.y); a[10]=f2bf(a2.z); a[11]=f2bf(a2.w);
    a[12]=f2bf(a3.x); a[13]=f2bf(a3.y); a[14]=f2bf(a3.z); a[15]=f2bf(a3.w);

    // B fragments: lane's column has 16 contiguous K values in w1t
    const int kb = k0 + hi * 16;
#pragma unroll
    for (int t = 0; t < 4; t++) {
      v16bf b = *(const v16bf*)&w1t[(t * 16 + mlo) * F_IN + kb];
      acc[t] = __builtin_amdgcn_wmma_f32_16x16x32_bf16(
          false, a, false, b, (short)0, acc[t], false, false);
    }
  }

  // bias + relu, stage hidden tile to LDS row-major for re-fragmentation
#pragma unroll
  for (int t = 0; t < 4; t++) {
    const int col = t * 16 + mlo;
#pragma unroll
    for (int i = 0; i < 8; i++) {
      float v = acc[t][i] + b1s[col];
      v = v > 0.f ? v : 0.f;
      hbuf[wave][(i + hi * 8) * HID + col] = f2bf(v);
    }
  }
  __syncthreads();   // all 128 threads participate (no early returns)

  // ---------------- GEMM 2: 16x64 @ 64x64 ----------------
  v8f acc2[4] = {v8f{}, v8f{}, v8f{}, v8f{}};
  const __bf16* hb = &hbuf[wave][0];
#pragma unroll
  for (int k0 = 0; k0 < HID; k0 += 32) {
    bf16x8 lo = *(const bf16x8*)(hb + mlo * HID + k0 + hi * 16);
    bf16x8 hp = *(const bf16x8*)(hb + mlo * HID + k0 + hi * 16 + 8);
    v16bf a = __builtin_shufflevector(lo, hp, 0,1,2,3,4,5,6,7,8,9,10,11,12,13,14,15);
    const int kb = k0 + hi * 16;
#pragma unroll
    for (int t = 0; t < 4; t++) {
      v16bf b = *(const v16bf*)&w2t[(t * 16 + mlo) * HID + kb];
      acc2[t] = __builtin_amdgcn_wmma_f32_16x16x32_bf16(
          false, a, false, b, (short)0, acc2[t], false, false);
    }
  }

  if (!valid) return;
#pragma unroll
  for (int t = 0; t < 4; t++) {
    const int col = t * 16 + mlo;
#pragma unroll
    for (int i = 0; i < 8; i++) {
      out[(row0 + i + hi * 8) * NCLS + col] = acc2[t][i] + b2s[col];
    }
  }
}

// =================================================================
// Kernel 2: zero a buffer (float4 granularity)
// =================================================================
__global__ void dagnn_zero(float4* __restrict__ p, long n4)
{
  long i = (long)blockIdx.x * blockDim.x + threadIdx.x;
  if (i < n4) p[i] = make_float4(0.f, 0.f, 0.f, 0.f);
}

// =================================================================
// Kernel 3: SpMM  y[r] += w[e] * x[s];  16 lanes per edge, 4 ch/lane
// =================================================================
__global__ void dagnn_spmm(const float* __restrict__ x, float* __restrict__ y,
                           const float* __restrict__ w,
                           const int* __restrict__ snd, const int* __restrict__ rcv,
                           long nE)
{
  long t = (long)blockIdx.x * blockDim.x + threadIdx.x;
  long e = t >> 4;
  int  g = (int)(t & 15);
  if (e >= nE) return;
  const int  s  = snd[e];
  const int  r  = rcv[e];
  const float we = w[e];
  float4 v = *(const float4*)(x + (long)s * NCLS + g * 4);
  float* yp = y + (long)r * NCLS + g * 4;
  atomicAdd(yp + 0, we * v.x);
  atomicAdd(yp + 1, we * v.y);
  atomicAdd(yp + 2, we * v.z);
  atomicAdd(yp + 3, we * v.w);
}

// =================================================================
// Kernel 4: gated accumulate:  out[n] (+)= sigmoid(x[n]·Wg + bg) * x[n]
// =================================================================
__global__ void dagnn_gate(const float* __restrict__ x,
                           const float* __restrict__ Wg, const float* __restrict__ bg,
                           float* __restrict__ out, int n, int init)
{
  int node = blockIdx.x * blockDim.x + threadIdx.x;
  if (node >= n) return;
  const float* xp = x + (long)node * NCLS;
  float dot = bg[0];
  float4 xv[16];
#pragma unroll
  for (int i = 0; i < 16; i++) {
    xv[i] = *(const float4*)(xp + i * 4);
    dot += xv[i].x * Wg[i*4+0] + xv[i].y * Wg[i*4+1]
         + xv[i].z * Wg[i*4+2] + xv[i].w * Wg[i*4+3];
  }
  const float sc = 1.f / (1.f + __expf(-dot));
  float* op = out + (long)node * NCLS;
#pragma unroll
  for (int i = 0; i < 16; i++) {
    float4 o = init ? make_float4(0.f,0.f,0.f,0.f) : *(const float4*)(op + i * 4);
    o.x += sc * xv[i].x;  o.y += sc * xv[i].y;
    o.z += sc * xv[i].z;  o.w += sc * xv[i].w;
    *(float4*)(op + i * 4) = o;
  }
}

// =================================================================
extern "C" void kernel_launch(void* const* d_in, const int* in_sizes, int n_in,
                              void* d_out, int out_size, void* d_ws, size_t ws_size,
                              hipStream_t stream) {
  const float* X   = (const float*)d_in[0];
  const float* ew  = (const float*)d_in[1];
  const float* W1  = (const float*)d_in[2];
  const float* b1  = (const float*)d_in[3];
  const float* W2  = (const float*)d_in[4];
  const float* b2  = (const float*)d_in[5];
  const float* Wg  = (const float*)d_in[6];
  const float* bg  = (const float*)d_in[7];
  const int*   snd = (const int*)d_in[8];
  const int*   rcv = (const int*)d_in[9];
  float* out = (float*)d_out;

  const int  n = in_sizes[0] / F_IN;     // 100000
  const long E = (long)in_sizes[1];      // 3200000

  float* buf0 = (float*)d_ws;
  float* buf1 = buf0 + (size_t)n * NCLS; // two 25.6 MB ping-pong hop buffers

  // logits -> buf0 (WMMA kernel)
  dagnn_mlp_wmma<<<(n + 63) / 64, 128, 0, stream>>>(X, W1, b1, W2, b2, buf0, n);
  // hop 0 contribution (initializes d_out)
  dagnn_gate<<<(n + 255) / 256, 256, 0, stream>>>(buf0, Wg, bg, out, n, 1);

  const long n4 = (long)n * (NCLS / 4);
  const long T  = E * 16;
  for (int k = 0; k < 20; k++) {
    dagnn_zero<<<(unsigned)((n4 + 255) / 256), 256, 0, stream>>>((float4*)buf1, n4);
    dagnn_spmm<<<(unsigned)((T + 255) / 256), 256, 0, stream>>>(buf0, buf1, ew, snd, rcv, E);
    dagnn_gate<<<(n + 255) / 256, 256, 0, stream>>>(buf1, Wg, bg, out, n, 0);
    float* tmp = buf0; buf0 = buf1; buf1 = tmp;
  }
}
